// Decoder_80711025427139
// MI455X (gfx1250) — compile-verified, matrix operand order checked by reading, not run on hardware
//
#include <hip/hip_runtime.h>

// ---------------- constants ----------------
#define B_    32
#define S_    512
#define EMB_  256
#define PRE_  256
#define RNN_  1024
#define ATT_  128
#define MEL_  80
#define T_    400
#define LF_   32
#define LK_   31
#define G4_   4096          // 4*RNN
#define NWG_  32            // persistent blocks (== B_)
#define TPB_  256           // 8 wave32 waves per block
#define NT_   (NWG_*TPB_)   // 8192 threads
#define NWAVE_ (NT_/32)     // 256 waves

typedef __bf16 bf16_t;
typedef __bf16 bf16x8 __attribute__((ext_vector_type(8)));
typedef __bf16 v16bf  __attribute__((ext_vector_type(16)));
typedef float  v8f    __attribute__((ext_vector_type(8)));

struct Params {
  // inputs (setup_inputs order)
  const float *emb, *tmels, *W1, *b1, *W2, *b2;
  const float *Wih_pre, *Whh_pre, *bih_pre, *bhh_pre;
  const float *Wq, *Wmem, *conv_w, *Wloc, *v;
  const float *Wih_post, *Whh_post, *bih_post, *bhh_post;
  const float *Wmel, *bmel, *Wstop, *bstop;
  float* out;
  // scratch
  unsigned* bar;
  bf16_t *wihpre_b, *whhpre_b, *wihpost_b, *whhpost_b, *wmem_b, *wloc_b, *emb_b;
  bf16_t *xpre_b, *hpre_b, *xpost_b, *hpost_b, *locT_b;
  float *p1all, *p2all, *memproj, *lp, *gates, *q, *energ, *a_w, *a_sum, *ctx;
  float *cPre, *hPre, *cPost, *hPost;
};

// ---------------- device helpers ----------------
__device__ __forceinline__ float sigm(float x) { return 1.f / (1.f + __expf(-x)); }

__device__ __forceinline__ unsigned hashu(unsigned x) {
  x ^= x >> 16; x *= 0x7feb352dU; x ^= x >> 15; x *= 0x846ca68bU; x ^= x >> 16; return x;
}
__device__ __forceinline__ float dropval(float v, float pdrop, unsigned h) {
  float keep = 1.f - pdrop;
  return ((h & 0xFFFFFFu) < (unsigned)(keep * 16777216.f)) ? v / keep : 0.f;
}

// device-scope split barrier: single monotonic counter (zeroed by init kernel each launch)
__device__ __forceinline__ void gbar(unsigned* bar, unsigned target) {
  __syncthreads();
  if (threadIdx.x == 0) {
    __threadfence();
    __hip_atomic_fetch_add(bar, 1u, __ATOMIC_RELEASE, __HIP_MEMORY_SCOPE_AGENT);
    while (__hip_atomic_load(bar, __ATOMIC_ACQUIRE, __HIP_MEMORY_SCOPE_AGENT) < target)
      __builtin_amdgcn_s_sleep(4);
  }
  __syncthreads();
  __threadfence();
}

// WMMA bf16 16x16x32, fp32 accumulate
__device__ __forceinline__ v8f wmma_bf16(v16bf a, v16bf b, v8f c) {
  return __builtin_amdgcn_wmma_f32_16x16x32_bf16(false, a, false, b, (short)0, c, false, false);
}

// A/B fragment load for one 16x32 (or 32x16) bf16 tile from row-major [16 rows, K] base.
// ISA 16-bit A layout: lane L(0..15) row M=L, K-halves [kb+koff, kb+koff+8) and
// [kb+16+koff, +8), koff = 8*(lane>=16)  ->  two contiguous 16B loads per lane.
__device__ __forceinline__ v16bf load_ab(const bf16_t* base, int K, int kb, int lane) {
  const bf16_t* ptr = base + (long)(lane & 15) * K + kb + ((lane >> 4) << 3);
  bf16x8 lo = *(const bf16x8*)ptr;
  bf16x8 hi = *(const bf16x8*)(ptr + 16);
  return __builtin_shufflevector(lo, hi, 0,1,2,3,4,5,6,7,8,9,10,11,12,13,14,15);
}

// gates GEMM: [32,Kx]@Wx^T + [32,Kh]@Wh^T -> gates[32,4096] (+bias). 512 tiles over 256 waves.
__device__ __forceinline__ void gemm_gates(const bf16_t* X, int Kx, const bf16_t* Wx,
                                           const bf16_t* H, int Kh, const bf16_t* Wh,
                                           const float* bi, const float* bh,
                                           float* gout, int wave, int lane) {
  for (int tile = wave; tile < 2 * (G4_ / 16); tile += NWAVE_) {
    int mt = tile & 1, nt = tile >> 1;
    const bf16_t* A0 = X  + (long)(mt * 16) * Kx;
    const bf16_t* B0 = Wx + (long)(nt * 16) * Kx;
    v8f acc = {0.f,0.f,0.f,0.f,0.f,0.f,0.f,0.f};
    for (int kb = 0; kb < Kx; kb += 32)
      acc = wmma_bf16(load_ab(A0, Kx, kb, lane), load_ab(B0, Kx, kb, lane), acc);
    const bf16_t* A1 = H  + (long)(mt * 16) * Kh;
    const bf16_t* B1 = Wh + (long)(nt * 16) * Kh;
    for (int kb = 0; kb < Kh; kb += 32)
      acc = wmma_bf16(load_ab(A1, Kh, kb, lane), load_ab(B1, Kh, kb, lane), acc);
    // C/D layout: VGPR r -> M = r + 8*(lane>=16), N = lane&15
    int n = nt * 16 + (lane & 15);
    float bias = bi[n] + bh[n];
    int mb = mt * 16 + ((lane >> 4) << 3);
    #pragma unroll
    for (int r = 0; r < 8; r++) gout[(long)(mb + r) * G4_ + n] = acc[r] + bias;
  }
}

// ---------------- init kernel 1: reset barrier, zero state, bf16 weights, PreNet L1 ----------------
__global__ __launch_bounds__(TPB_) void k_init1(Params p) {
  long gid = (long)blockIdx.x * TPB_ + threadIdx.x;
  long gsz = (long)gridDim.x * TPB_;
  if (gid == 0) *p.bar = 0u;
  for (long i = gid; i < B_ * RNN_; i += gsz) { p.cPre[i]=0.f; p.hPre[i]=0.f; p.cPost[i]=0.f; p.hPost[i]=0.f; }
  for (long i = gid; i < B_ * EMB_; i += gsz) p.ctx[i] = 0.f;
  for (long i = gid; i < B_ * S_;   i += gsz) { p.a_w[i]=0.f; p.a_sum[i]=0.f; }
  for (long i = gid; i < (long)G4_ * 512;  i += gsz) p.wihpre_b[i]  = (bf16_t)p.Wih_pre[i];
  for (long i = gid; i < (long)G4_ * 1024; i += gsz) p.whhpre_b[i]  = (bf16_t)p.Whh_pre[i];
  for (long i = gid; i < (long)G4_ * 1280; i += gsz) p.wihpost_b[i] = (bf16_t)p.Wih_post[i];
  for (long i = gid; i < (long)G4_ * 1024; i += gsz) p.whhpost_b[i] = (bf16_t)p.Whh_post[i];
  for (long i = gid; i < ATT_ * EMB_; i += gsz) p.wmem_b[i] = (bf16_t)p.Wmem[i];
  for (long i = gid; i < ATT_ * LF_;  i += gsz) p.wloc_b[i] = (bf16_t)p.Wloc[i];
  for (long i = gid; i < (long)B_ * S_ * EMB_; i += gsz) p.emb_b[i] = (bf16_t)p.emb[i];
  // p1_all[t][b][j] = dropout(relu(mel_prev @ W1^T + b1), 0.5)
  for (long i = gid; i < (long)T_ * B_ * PRE_; i += gsz) {
    int j = (int)(i & 255); int b = (int)((i >> 8) & 31); int t = (int)(i >> 13);
    float acc = p.b1[j];
    if (t > 0) {
      const float* mp = p.tmels + ((long)b * T_ + (t - 1)) * MEL_;
      const float* w  = p.W1 + (long)j * MEL_;
      for (int m = 0; m < MEL_; m++) acc += mp[m] * w[m];
    }
    acc = fmaxf(acc, 0.f);
    p.p1all[i] = dropval(acc, 0.5f, hashu(0xA511E91u ^ (unsigned)i ^ (unsigned)(t * 0x9E3779B9u)));
  }
}

// ---------------- init kernel 2: PreNet L2 + mem_proj (WMMA) ----------------
__global__ __launch_bounds__(TPB_) void k_init2(Params p) {
  long gid = (long)blockIdx.x * TPB_ + threadIdx.x;
  long gsz = (long)gridDim.x * TPB_;
  for (long i = gid; i < (long)T_ * B_ * PRE_; i += gsz) {
    int j = (int)(i & 255); long tb = i >> 8;
    const float* p1 = p.p1all + tb * 256;
    const float* w  = p.W2 + (long)j * PRE_;
    float acc = p.b2[j];
    for (int k = 0; k < PRE_; k++) acc += p1[k] * w[k];
    acc = fmaxf(acc, 0.f);
    p.p2all[i] = dropval(acc, 0.5f, hashu(0xB522F17u ^ (unsigned)i));
  }
  // mem_proj[16384,128] = emb[16384,256] @ Wmem^T   (1024x8 tiles, K=256 -> 8 WMMAs each)
  int wave = (int)(gid >> 5), lane = threadIdx.x & 31;
  int nwaves = (int)(gsz >> 5);
  for (int tile = wave; tile < (B_ * S_ / 16) * (ATT_ / 16); tile += nwaves) {
    int mt = tile >> 3, ntl = tile & 7;
    const bf16_t* A  = p.emb_b  + (long)mt * 16 * EMB_;
    const bf16_t* Bm = p.wmem_b + (long)ntl * 16 * EMB_;
    v8f acc = {0.f,0.f,0.f,0.f,0.f,0.f,0.f,0.f};
    for (int kb = 0; kb < EMB_; kb += 32)
      acc = wmma_bf16(load_ab(A, EMB_, kb, lane), load_ab(Bm, EMB_, kb, lane), acc);
    int n = ntl * 16 + (lane & 15);
    int mb = mt * 16 + ((lane >> 4) << 3);
    #pragma unroll
    for (int r = 0; r < 8; r++) p.memproj[(long)(mb + r) * ATT_ + n] = acc[r];
  }
}

// ---------------- persistent decoder: 400 steps, 6 global barriers per step ----------------
__global__ __launch_bounds__(TPB_) void k_decoder(Params p) {
  const int tid = threadIdx.x, bid = blockIdx.x;
  const int gtid = bid * TPB_ + tid;
  const int wave = gtid >> 5, lane = tid & 31;
  __shared__ float red[TPB_];
  unsigned bt = 0;

  for (int t = 0; t < T_; t++) {
    // ---- Phase A: build Xpre/Hpre bf16 + location conv -> locT (all blocks) ----
    for (int i = gtid; i < B_ * (PRE_ + EMB_); i += NT_) {
      int b = i >> 9, k = i & 511;
      float val = (k < PRE_) ? p.p2all[((long)t * B_ + b) * PRE_ + k]
                             : p.ctx[b * EMB_ + (k - PRE_)];
      p.xpre_b[i] = (bf16_t)val;
    }
    for (int i = gtid; i < B_ * RNN_; i += NT_) p.hpre_b[i] = (bf16_t)p.cPre[i];
    for (int i = gtid; i < B_ * LF_ * S_; i += NT_) {
      int s = i & 511, f = (i >> 9) & 31, b = i >> 14;
      const float* w0 = p.conv_w + (f * 2 + 0) * LK_;
      const float* w1 = p.conv_w + (f * 2 + 1) * LK_;
      const float* c0 = p.a_sum + b * S_;
      const float* c1 = p.a_w   + b * S_;
      float acc = 0.f;
      for (int k = 0; k < LK_; k++) {
        int sp = s + k - LK_ / 2;
        if (sp >= 0 && sp < S_) acc += c0[sp] * w0[k] + c1[sp] * w1[k];
      }
      p.locT_b[(long)(b * S_ + s) * LF_ + f] = (bf16_t)acc;
    }
    bt += NWG_; gbar(p.bar, bt);

    // ---- Phase B: gates_pre GEMM + loc_proj GEMM (WMMA, all waves) ----
    gemm_gates(p.xpre_b, PRE_ + EMB_, p.wihpre_b, p.hpre_b, RNN_, p.whhpre_b,
               p.bih_pre, p.bhh_pre, p.gates, wave, lane);
    for (int tile = wave; tile < (B_ * S_ / 16) * (ATT_ / 16); tile += NWAVE_) {
      int mt = tile >> 3, ntl = tile & 7;
      const bf16_t* A  = p.locT_b + (long)mt * 16 * LF_;
      const bf16_t* Bm = p.wloc_b + (long)ntl * 16 * LF_;
      v8f acc = {0.f,0.f,0.f,0.f,0.f,0.f,0.f,0.f};
      acc = wmma_bf16(load_ab(A, LF_, 0, lane), load_ab(Bm, LF_, 0, lane), acc);
      int n = ntl * 16 + (lane & 15);
      int mb = mt * 16 + ((lane >> 4) << 3);
      #pragma unroll
      for (int r = 0; r < 8; r++) p.lp[(long)(mb + r) * ATT_ + n] = acc[r];
    }
    bt += NWG_; gbar(p.bar, bt);

    // ---- Phase C1 (per-b): pre LSTM cell (+dropout 0.1) then q = h_pre @ Wq^T ----
    {
      const int b = bid;
      const float* g = p.gates + (long)b * G4_;
      for (int j = tid; j < RNN_; j += TPB_) {
        float gi = g[j], gf = g[RNN_ + j], gg = g[2 * RNN_ + j], go = g[3 * RNN_ + j];
        float cin = p.hPre[b * RNN_ + j];                 // (swapped-state convention)
        float cn = sigm(gf) * cin + sigm(gi) * tanhf(gg);
        float hn = sigm(go) * tanhf(cn);
        hn = dropval(hn, 0.1f, hashu(0xC0FFEEu ^ (unsigned)(t * 0x9E3779B9u) ^ (unsigned)(b * RNN_ + j)));
        p.cPre[b * RNN_ + j] = hn;
        p.hPre[b * RNN_ + j] = cn;
      }
      __syncthreads();
      if (tid < ATT_) {
        const float* h = p.hPre + (long)b * RNN_;
        const float* w = p.Wq + (long)tid * RNN_;
        float s = 0.f;
        for (int k = 0; k < RNN_; k++) s += h[k] * w[k];
        p.q[b * ATT_ + tid] = s;
      }
    }
    bt += NWG_; gbar(p.bar, bt);

    // ---- Phase C2: energies (all blocks) ----
    for (int i = gtid; i < B_ * S_; i += NT_) {
      int b = i >> 9;
      const float* qq = p.q + b * ATT_;
      const float* mp = p.memproj + (long)i * ATT_;
      const float* lpp = p.lp + (long)i * ATT_;
      float e = 0.f;
      for (int a = 0; a < ATT_; a++) e += p.v[a] * tanhf(qq[a] + mp[a] + lpp[a]);
      p.energ[i] = e;
    }
    bt += NWG_; gbar(p.bar, bt);

    // ---- Phase C3 (per-b): softmax, context, a_sum, build Xpost/Hpost bf16 ----
    {
      const int b = bid;
      float m = -1e30f;
      for (int s = tid; s < S_; s += TPB_) m = fmaxf(m, p.energ[b * S_ + s]);
      red[tid] = m; __syncthreads();
      for (int o = TPB_ / 2; o > 0; o >>= 1) { if (tid < o) red[tid] = fmaxf(red[tid], red[tid + o]); __syncthreads(); }
      float mx = red[0]; __syncthreads();
      float sum = 0.f;
      for (int s = tid; s < S_; s += TPB_) { float e = __expf(p.energ[b * S_ + s] - mx); p.a_w[b * S_ + s] = e; sum += e; }
      red[tid] = sum; __syncthreads();
      for (int o = TPB_ / 2; o > 0; o >>= 1) { if (tid < o) red[tid] += red[tid + o]; __syncthreads(); }
      float inv = 1.f / red[0];
      __syncthreads();
      for (int s = tid; s < S_; s += TPB_) { float w = p.a_w[b * S_ + s] * inv; p.a_w[b * S_ + s] = w; p.a_sum[b * S_ + s] += w; }
      __syncthreads();
      for (int e = tid; e < EMB_; e += TPB_) {
        const float* aw = p.a_w + b * S_;
        const float* em = p.emb + (long)b * S_ * EMB_ + e;
        float c = 0.f;
        for (int s = 0; s < S_; s++) c += aw[s] * em[(long)s * EMB_];
        p.ctx[b * EMB_ + e] = c;
      }
      __syncthreads();
      for (int k = tid; k < RNN_ + EMB_; k += TPB_)
        p.xpost_b[(long)b * (RNN_ + EMB_) + k] =
            (bf16_t)((k < RNN_) ? p.hPre[b * RNN_ + k] : p.ctx[b * EMB_ + k - RNN_]);
      for (int k = tid; k < RNN_; k += TPB_)
        p.hpost_b[b * RNN_ + k] = (bf16_t)p.cPost[b * RNN_ + k];
    }
    bt += NWG_; gbar(p.bar, bt);

    // ---- Phase H: gates_post GEMM (WMMA, all waves) ----
    gemm_gates(p.xpost_b, RNN_ + EMB_, p.wihpost_b, p.hpost_b, RNN_, p.whhpost_b,
               p.bih_post, p.bhh_post, p.gates, wave, lane);
    bt += NWG_; gbar(p.bar, bt);

    // ---- Phase I (per-b): post LSTM cell (+dropout), mel/stop projections ----
    {
      const int b = bid;
      const float* g = p.gates + (long)b * G4_;
      for (int j = tid; j < RNN_; j += TPB_) {
        float gi = g[j], gf = g[RNN_ + j], gg = g[2 * RNN_ + j], go = g[3 * RNN_ + j];
        float cin = p.hPost[b * RNN_ + j];
        float cn = sigm(gf) * cin + sigm(gi) * tanhf(gg);
        float hn = sigm(go) * tanhf(cn);
        hn = dropval(hn, 0.1f, hashu(0xD00D1EU ^ (unsigned)(t * 0x85EBCA77u) ^ (unsigned)(b * RNN_ + j)));
        p.cPost[b * RNN_ + j] = hn;
        p.hPost[b * RNN_ + j] = cn;
      }
      __syncthreads();
      if (tid < MEL_) {
        const float* w = p.Wmel + (long)tid * (RNN_ + EMB_);
        float s = p.bmel[tid];
        for (int k = 0; k < RNN_; k++) s += p.hPost[b * RNN_ + k] * w[k];
        for (int k = 0; k < EMB_; k++) s += p.ctx[b * EMB_ + k] * w[RNN_ + k];
        p.out[((long)b * T_ + t) * MEL_ + tid] = sigm(s);
      }
      if (tid == 128) {
        const float* w = p.Wstop;
        float s = p.bstop[0];
        for (int k = 0; k < RNN_; k++) s += p.hPost[b * RNN_ + k] * w[k];
        for (int k = 0; k < EMB_; k++) s += p.ctx[b * EMB_ + k] * w[RNN_ + k];
        p.out[(long)B_ * T_ * MEL_ + (long)b * T_ + t] = sigm(s);
      }
    }
    // No global barrier needed before next Phase A (proof: any block entering
    // Phase B(t+1) has seen all blocks pass barrier-A(t+1), which program-order
    // implies all blocks finished Phase I(t); Phase A writes nothing Phase I reads).
  }
}

// ---------------- host launcher ----------------
extern "C" void kernel_launch(void* const* d_in, const int* in_sizes, int n_in,
                              void* d_out, int out_size, void* d_ws, size_t ws_size,
                              hipStream_t stream) {
  (void)in_sizes; (void)n_in; (void)out_size; (void)ws_size;
  Params p{};
  p.emb      = (const float*)d_in[0];
  p.tmels    = (const float*)d_in[1];
  p.W1       = (const float*)d_in[2];
  p.b1       = (const float*)d_in[3];
  p.W2       = (const float*)d_in[4];
  p.b2       = (const float*)d_in[5];
  p.Wih_pre  = (const float*)d_in[6];
  p.Whh_pre  = (const float*)d_in[7];
  p.bih_pre  = (const float*)d_in[8];
  p.bhh_pre  = (const float*)d_in[9];
  p.Wq       = (const float*)d_in[10];
  p.Wmem     = (const float*)d_in[11];
  p.conv_w   = (const float*)d_in[12];
  p.Wloc     = (const float*)d_in[13];
  p.v        = (const float*)d_in[14];
  p.Wih_post = (const float*)d_in[15];
  p.Whh_post = (const float*)d_in[16];
  p.bih_post = (const float*)d_in[17];
  p.bhh_post = (const float*)d_in[18];
  p.Wmel     = (const float*)d_in[19];
  p.bmel     = (const float*)d_in[20];
  p.Wstop    = (const float*)d_in[21];
  p.bstop    = (const float*)d_in[22];
  p.out      = (float*)d_out;

  char* w = (char*)d_ws;
  auto take = [&](size_t bytes) -> char* {
    char* r = w; w += (bytes + 255) & ~(size_t)255; return r;
  };
  p.bar       = (unsigned*)take(256);
  p.wihpre_b  = (bf16_t*)take((size_t)G4_ * 512  * 2);
  p.whhpre_b  = (bf16_t*)take((size_t)G4_ * 1024 * 2);
  p.wihpost_b = (bf16_t*)take((size_t)G4_ * 1280 * 2);
  p.whhpost_b = (bf16_t*)take((size_t)G4_ * 1024 * 2);
  p.wmem_b    = (bf16_t*)take((size_t)ATT_ * EMB_ * 2);
  p.wloc_b    = (bf16_t*)take((size_t)ATT_ * LF_ * 2);
  p.emb_b     = (bf16_t*)take((size_t)B_ * S_ * EMB_ * 2);
  p.xpre_b    = (bf16_t*)take((size_t)B_ * (PRE_ + EMB_) * 2);
  p.hpre_b    = (bf16_t*)take((size_t)B_ * RNN_ * 2);
  p.xpost_b   = (bf16_t*)take((size_t)B_ * (RNN_ + EMB_) * 2);
  p.hpost_b   = (bf16_t*)take((size_t)B_ * RNN_ * 2);
  p.locT_b    = (bf16_t*)take((size_t)B_ * S_ * LF_ * 2);
  p.p1all     = (float*)take((size_t)T_ * B_ * PRE_ * 4);
  p.p2all     = (float*)take((size_t)T_ * B_ * PRE_ * 4);
  p.memproj   = (float*)take((size_t)B_ * S_ * ATT_ * 4);
  p.lp        = (float*)take((size_t)B_ * S_ * ATT_ * 4);
  p.gates     = (float*)take((size_t)B_ * G4_ * 4);
  p.q         = (float*)take((size_t)B_ * ATT_ * 4);
  p.energ     = (float*)take((size_t)B_ * S_ * 4);
  p.a_w       = (float*)take((size_t)B_ * S_ * 4);
  p.a_sum     = (float*)take((size_t)B_ * S_ * 4);
  p.ctx       = (float*)take((size_t)B_ * EMB_ * 4);
  p.cPre      = (float*)take((size_t)B_ * RNN_ * 4);
  p.hPre      = (float*)take((size_t)B_ * RNN_ * 4);
  p.cPost     = (float*)take((size_t)B_ * RNN_ * 4);
  p.hPost     = (float*)take((size_t)B_ * RNN_ * 4);

  k_init1<<<dim3(1024), dim3(TPB_), 0, stream>>>(p);
  k_init2<<<dim3(1024), dim3(TPB_), 0, stream>>>(p);
  k_decoder<<<dim3(NWG_), dim3(TPB_), 0, stream>>>(p);
}
